// RNNModel_O_44186623541343
// MI455X (gfx1250) — compile-verified, matrix operand order checked by reading, not run on hardware
//
#include <hip/hip_runtime.h>

// ---------------------------------------------------------------------------
// Projected LSTM (LSTMO), 2 layers, MI455X (gfx1250, wave32, WMMA bf16).
// B=32, T=512, I=512, H=1024 (cell), R=512 (projection). Output [B,T,R] f32.
//
// v2: persistent-kernel design. One kernel launch per layer; 8 workgroups
// (64 waves) loop over all T timesteps internally:
//   phase A (all 64 waves) : gates = [h_prev | x_t] @ [W_hh | W_ih]^T + b
//                            -> sigmoid/tanh, cell update, m = o*tanh(c)
//   phase B (32 waves)     : h_t = m @ Wp^T  (bf16 recurrent + optional f32 out)
// separated by a device-scope grid barrier (8 blocks, all resident).
// Weights (~18 MB bf16) stay L2-resident across the whole sequence; the
// sequential chain pays zero kernel-launch overhead.
// ---------------------------------------------------------------------------

typedef __bf16 bf16_t;
typedef __attribute__((ext_vector_type(16))) __bf16 v16bf;
typedef __attribute__((ext_vector_type(8)))  float  v8f;
typedef __attribute__((ext_vector_type(4)))  unsigned int uint4v;

#define LSTM_B 32
#define LSTM_T 512
#define LSTM_I 512
#define LSTM_H 1024
#define LSTM_R 512
#define LSTM_G (4 * LSTM_H)   // 4096 gate width

// ---------------- helpers ----------------

__device__ __forceinline__ bf16_t to_bf16(float f) {
    unsigned int u = __float_as_uint(f);
    u += 0x7FFFu + ((u >> 16) & 1u);            // round-to-nearest-even
    unsigned short s = (unsigned short)(u >> 16);
    bf16_t r;
    __builtin_memcpy(&r, &s, 2);
    return r;
}

__device__ __forceinline__ float sigmoidf_fast(float x) {
    return 1.0f / (1.0f + __expf(-x));
}
__device__ __forceinline__ float tanhf_fast(float x) {
    return 1.0f - 2.0f / (__expf(2.0f * x) + 1.0f);
}

union FragU { v16bf v; uint4v u[2]; };

// Load one wave32 WMMA bf16 fragment slice for this lane.
// Layout (ISA 16-bit A 16x32 / B 32x16, identical per-lane packing):
//   lane 0-15 : K in [k0 .. k0+7]  and [k0+16 .. k0+23]
//   lane 16-31: K in [k0+8..k0+15] and [k0+24 .. k0+31]
__device__ __forceinline__ v16bf load_frag(const bf16_t* __restrict__ row,
                                           int k0, int half) {
    FragU f;
    f.u[0] = *reinterpret_cast<const uint4v*>(row + k0 + half * 8);
    f.u[1] = *reinterpret_cast<const uint4v*>(row + k0 + 16 + half * 8);
    return f.v;
}

__device__ __forceinline__ v8f wmma_bf16(v16bf a, v16bf b, v8f c) {
    return __builtin_amdgcn_wmma_f32_16x16x32_bf16(
        /*neg_a=*/false, a, /*neg_b=*/false, b,
        /*c_mod=*/(short)0, c, /*reuse_a=*/false, /*reuse_b=*/false);
}

// Device-scope grid barrier (counter + generation). Safe: gridDim.x == 8,
// all blocks co-resident. No WMMA inside divergent region.
__device__ __forceinline__ void grid_sync(unsigned int* cnt, unsigned int* gen,
                                          unsigned int nb) {
    __threadfence();
    __syncthreads();
    if (threadIdx.x == 0) {
        unsigned int g = __hip_atomic_load(gen, __ATOMIC_RELAXED,
                                           __HIP_MEMORY_SCOPE_AGENT);
        unsigned int a = __hip_atomic_fetch_add(cnt, 1u, __ATOMIC_ACQ_REL,
                                                __HIP_MEMORY_SCOPE_AGENT);
        if (a == nb - 1u) {
            __hip_atomic_store(cnt, 0u, __ATOMIC_RELAXED,
                               __HIP_MEMORY_SCOPE_AGENT);
            __hip_atomic_store(gen, g + 1u, __ATOMIC_RELEASE,
                               __HIP_MEMORY_SCOPE_AGENT);
        } else {
            while (__hip_atomic_load(gen, __ATOMIC_ACQUIRE,
                                     __HIP_MEMORY_SCOPE_AGENT) == g) {
                __builtin_amdgcn_s_sleep(2);
            }
        }
    }
    __syncthreads();
    __threadfence();
}

// ---------------- kernels ----------------

// f32 -> bf16 elementwise conversion
__global__ void __launch_bounds__(256)
f32_to_bf16_k(const float* __restrict__ in, bf16_t* __restrict__ out, int n) {
    int i = blockIdx.x * blockDim.x + threadIdx.x;
    if (i < n) out[i] = to_bf16(in[i]);
}

// Persistent layer kernel: grid = 8 blocks x 256 threads = 64 waves.
// Wave j (= blockIdx*8 + waveInBlock) owns hidden cols [16j, 16j+16) of all
// four gates in phase A; even-j waves own projection cols [16*(j/2), +16) in
// phase B (spread 4 proj waves per block).
__global__ void __launch_bounds__(256)
lstmo_layer(const bf16_t* __restrict__ xbase,   // time-major input rows
            long xt_tstride,                    // elems between timesteps
            long xt_bstride,                    // elems between batch rows
            const bf16_t* __restrict__ Whh,     // [4096, 512] bf16
            const bf16_t* __restrict__ Wih,     // [4096, 512] bf16
            const float*  __restrict__ bias,    // [4096] f32
            const bf16_t* __restrict__ Wp,      // [512, 1024] bf16
            float*  __restrict__ c,             // [32, 1024] f32 (zeroed)
            bf16_t* __restrict__ m,             // [32, 1024] bf16 scratch
            bf16_t* __restrict__ hseq,          // h slots, stride hseq_tstride
            long hseq_tstride,                  // 0 => single ping buffer
            const bf16_t* __restrict__ hzero,   // [32, 512] zeros
            float* __restrict__ outf,           // optional f32 out base
            long outf_tstride, long outf_bstride,
            unsigned int* __restrict__ bar_cnt,
            unsigned int* __restrict__ bar_gen,
            int T)
{
    const int lane = threadIdx.x & 31;
    const int wave = threadIdx.x >> 5;
    const int j    = blockIdx.x * 8 + wave;     // 0..63
    const int half = lane >> 4;
    const int lm   = lane & 15;
    const int hc0  = j * 16;                    // gate-col base [0,1024)
    const unsigned int nb = gridDim.x;

    // Gate B-row pointers are t-invariant: hoist out of the time loop.
    const bf16_t* browH[4];
    const bf16_t* browI[4];
#pragma unroll
    for (int g = 0; g < 4; ++g) {
        browH[g] = Whh + (size_t)(g * LSTM_H + hc0 + lm) * LSTM_R;
        browI[g] = Wih + (size_t)(g * LSTM_H + hc0 + lm) * LSTM_I;
    }
    const int col = hc0 + lm;                   // hidden index [0,1024)
    const float bi = bias[0 * LSTM_H + col];
    const float bf = bias[1 * LSTM_H + col];
    const float bg = bias[2 * LSTM_H + col];
    const float bo = bias[3 * LSTM_H + col];

    // Projection constants (even-j waves only).
    const int  pj    = j >> 1;                  // 0..31
    const int  pcol  = pj * 16 + lm;            // [0, 512)
    const bf16_t* pbrow = Wp + (size_t)pcol * LSTM_H;

    for (int t = 0; t < T; ++t) {
        const bf16_t* hp = (t == 0)
            ? hzero : (hseq + (size_t)(t - 1) * hseq_tstride);
        const bf16_t* xt = xbase + (size_t)t * xt_tstride;

        // ================= phase A: gates (all 64 waves) =================
        {
            v8f acc[2][4];
#pragma unroll
            for (int a = 0; a < 2; ++a)
#pragma unroll
                for (int g = 0; g < 4; ++g) acc[a][g] = 0.0f;

            // recurrent part: h_prev @ W_hh^T (K = 512)
            {
                const bf16_t* arow0 = hp + (size_t)lm * LSTM_R;
                const bf16_t* arow1 = hp + (size_t)(16 + lm) * LSTM_R;
                for (int k = 0; k < LSTM_R; k += 32) {
                    v16bf a0 = load_frag(arow0, k, half);
                    v16bf a1 = load_frag(arow1, k, half);
#pragma unroll
                    for (int g = 0; g < 4; ++g) {
                        v16bf bb = load_frag(browH[g], k, half);
                        acc[0][g] = wmma_bf16(a0, bb, acc[0][g]);
                        acc[1][g] = wmma_bf16(a1, bb, acc[1][g]);
                    }
                }
            }
            // input part: x_t @ W_ih^T (K = 512)
            {
                const bf16_t* arow0 = xt + (size_t)lm * xt_bstride;
                const bf16_t* arow1 = xt + (size_t)(16 + lm) * xt_bstride;
                for (int k = 0; k < LSTM_I; k += 32) {
                    v16bf a0 = load_frag(arow0, k, half);
                    v16bf a1 = load_frag(arow1, k, half);
#pragma unroll
                    for (int g = 0; g < 4; ++g) {
                        v16bf bb = load_frag(browI[g], k, half);
                        acc[0][g] = wmma_bf16(a0, bb, acc[0][g]);
                        acc[1][g] = wmma_bf16(a1, bb, acc[1][g]);
                    }
                }
            }
            // epilogue: bias, nonlinearities, cell update, m
#pragma unroll
            for (int mt = 0; mt < 2; ++mt) {
#pragma unroll
                for (int r = 0; r < 8; ++r) {
                    const int b = mt * 16 + half * 8 + r;   // batch row
                    const float ig = sigmoidf_fast(acc[mt][0][r] + bi);
                    const float fg = sigmoidf_fast(acc[mt][1][r] + bf);
                    const float gg = tanhf_fast  (acc[mt][2][r] + bg);
                    const float og = sigmoidf_fast(acc[mt][3][r] + bo);
                    const size_t idx = (size_t)b * LSTM_H + col;
                    const float ct = fg * c[idx] + ig * gg;
                    c[idx] = ct;                   // same wave every t: no sync
                    m[idx] = to_bf16(og * tanhf_fast(ct));
                }
            }
        }

        grid_sync(bar_cnt, bar_gen, nb);        // m visible to all proj waves

        // ============== phase B: projection (32 even-j waves) =============
        if ((j & 1) == 0) {
            v8f acc0 = 0.0f, acc1 = 0.0f;
            const bf16_t* arow0 = m + (size_t)lm * LSTM_H;
            const bf16_t* arow1 = m + (size_t)(16 + lm) * LSTM_H;
            for (int k = 0; k < LSTM_H; k += 32) {
                v16bf a0 = load_frag(arow0, k, half);
                v16bf a1 = load_frag(arow1, k, half);
                v16bf bb = load_frag(pbrow, k, half);
                acc0 = wmma_bf16(a0, bb, acc0);
                acc1 = wmma_bf16(a1, bb, acc1);
            }
            bf16_t* hb = hseq + (size_t)t * hseq_tstride;
            float*  hf = outf ? (outf + (size_t)t * outf_tstride) : (float*)0;
#pragma unroll
            for (int r = 0; r < 8; ++r) {
                const int b0 = half * 8 + r;
                const int b1 = 16 + half * 8 + r;
                const float v0 = acc0[r];
                const float v1 = acc1[r];
                hb[(size_t)b0 * LSTM_R + pcol] = to_bf16(v0);
                hb[(size_t)b1 * LSTM_R + pcol] = to_bf16(v1);
                if (hf) {
                    hf[(size_t)b0 * outf_bstride + pcol] = v0;
                    hf[(size_t)b1 * outf_bstride + pcol] = v1;
                }
            }
        }

        grid_sync(bar_cnt, bar_gen, nb);        // h_t visible before t+1
    }
}

// ---------------- host launcher ----------------

extern "C" void kernel_launch(void* const* d_in, const int* in_sizes, int n_in,
                              void* d_out, int out_size, void* d_ws, size_t ws_size,
                              hipStream_t stream) {
    (void)in_sizes; (void)n_in; (void)out_size; (void)ws_size;

    const float* x    = (const float*)d_in[0];
    const float* Wih0 = (const float*)d_in[1];
    const float* Whh0 = (const float*)d_in[2];
    const float* b0   = (const float*)d_in[3];
    const float* Wp0  = (const float*)d_in[4];
    const float* Wih1 = (const float*)d_in[5];
    const float* Whh1 = (const float*)d_in[6];
    const float* b1   = (const float*)d_in[7];
    const float* Wp1  = (const float*)d_in[8];
    float* out = (float*)d_out;

    // ---- workspace carve-up ----
    char* ws = (char*)d_ws;
    const size_t N_X   = (size_t)LSTM_B * LSTM_T * LSTM_I;     // 8,388,608
    const size_t N_WG  = (size_t)LSTM_G * LSTM_I;              // 2,097,152 (I==R)
    const size_t N_WP  = (size_t)LSTM_R * LSTM_H;              // 524,288
    const size_t N_SEQ = (size_t)LSTM_T * LSTM_B * LSTM_R;     // 8,388,608
    const size_t N_C   = (size_t)LSTM_B * LSTM_H;              // 32,768
    const size_t N_HP  = (size_t)LSTM_B * LSTM_R;              // 16,384

    size_t off = 0;
    bf16_t* x_b    = (bf16_t*)(ws + off); off += N_X   * 2;
    bf16_t* Wih0_b = (bf16_t*)(ws + off); off += N_WG  * 2;
    bf16_t* Whh0_b = (bf16_t*)(ws + off); off += N_WG  * 2;
    bf16_t* Wp0_b  = (bf16_t*)(ws + off); off += N_WP  * 2;
    bf16_t* Wih1_b = (bf16_t*)(ws + off); off += N_WG  * 2;
    bf16_t* Whh1_b = (bf16_t*)(ws + off); off += N_WG  * 2;
    bf16_t* Wp1_b  = (bf16_t*)(ws + off); off += N_WP  * 2;
    bf16_t* h0seq  = (bf16_t*)(ws + off); off += N_SEQ * 2;    // [T, B, R]
    float*  cbuf   = (float*) (ws + off); off += N_C   * 4;
    bf16_t* mbuf   = (bf16_t*)(ws + off); off += N_C   * 2;
    bf16_t* hprev1 = (bf16_t*)(ws + off); off += N_HP  * 2;
    bf16_t* hzero  = (bf16_t*)(ws + off); off += N_HP  * 2;
    off = (off + 255) & ~(size_t)255;
    unsigned int* bar = (unsigned int*)(ws + off); off += 256; // cnt, gen

    // ---- fp32 -> bf16 conversions (every call: deterministic) ----
    auto conv = [&](const float* src, bf16_t* dst, size_t n) {
        f32_to_bf16_k<<<(unsigned)((n + 255) / 256), 256, 0, stream>>>(src, dst, (int)n);
    };
    conv(x,    x_b,    N_X);
    conv(Wih0, Wih0_b, N_WG);
    conv(Whh0, Whh0_b, N_WG);
    conv(Wp0,  Wp0_b,  N_WP);
    conv(Wih1, Wih1_b, N_WG);
    conv(Whh1, Whh1_b, N_WG);
    conv(Wp1,  Wp1_b,  N_WP);

    hipMemsetAsync(hzero, 0, N_HP * 2, stream);
    hipMemsetAsync(bar,   0, 8,        stream);

    // ---- layer 0 (persistent: whole sequence in one launch) ----
    hipMemsetAsync(cbuf, 0, N_C * 4, stream);
    lstmo_layer<<<8, 256, 0, stream>>>(
        x_b, (long)LSTM_I, (long)LSTM_T * LSTM_I,           // x[b][t][:] rows
        Whh0_b, Wih0_b, b0, Wp0_b,
        cbuf, mbuf,
        h0seq, (long)LSTM_B * LSTM_R,                       // h slots per t
        hzero,
        (float*)nullptr, 0L, 0L,
        bar, bar + 1, LSTM_T);

    // ---- layer 1 ----
    hipMemsetAsync(cbuf, 0, N_C * 4, stream);
    lstmo_layer<<<8, 256, 0, stream>>>(
        h0seq, (long)LSTM_B * LSTM_R, (long)LSTM_R,         // h0seq[t][b][:]
        Whh1_b, Wih1_b, b1, Wp1_b,
        cbuf, mbuf,
        hprev1, 0L,                                         // single ping buffer
        hzero,
        out, (long)LSTM_R, (long)LSTM_T * LSTM_R,           // out[b][t][:]
        bar, bar + 1, LSTM_T);
}
